// SparseNeuralNetwork_57440892617097
// MI455X (gfx1250) — matrix operand order repaired and checked by reading, not compiled, not run on hardware
//
#include <hip/hip_runtime.h>

typedef __attribute__((ext_vector_type(16))) __bf16 v16bf;
typedef __attribute__((ext_vector_type(8)))  float  v8f;

#define IN_DIM  16
#define OUT_DIM 16
#define H0      16
#define H1      8

// Async memory->LDS copy of a contiguous f32 range using the gfx1250
// ASYNCcnt-tracked DMA path (global_load_async_to_lds_b128).
// Low 32 bits of a generic LDS pointer == LDS byte offset (flat aperture rule).
__device__ inline void async_stage_f32(const float* __restrict__ g,
                                       float* lds_dst, int bytes_total, int t) {
    unsigned           lbase = (unsigned)(uintptr_t)lds_dst;
    unsigned long long gbase = (unsigned long long)(uintptr_t)g;
    for (int off = t * 16; off < bytes_total; off += 256 * 16) {
        unsigned           lofs = lbase + off;
        unsigned long long ga   = gbase + off;
        asm volatile("global_load_async_to_lds_b128 %0, %1, off"
                     :: "v"(lofs), "v"(ga) : "memory");
    }
}

// One wave (32 lanes) handles a 16-row batch tile.
// Lanes L and L+16 mirror batch row m = L&15 (A-matrix lane-half layout).
// Each WMMA packs TWO (j, i) blocks: block j  -> K 0..15  feeding N 0..7,
//                                    block j+1-> K 16..31 feeding N 8..15.
__global__ __launch_bounds__(256) void sparse_mlp_wmma(
    const float* __restrict__ x,
    const float* __restrict__ W0,
    const float* __restrict__ b0,
    const float* __restrict__ W1,
    const float* __restrict__ b1,
    const float* __restrict__ W2,
    const float* __restrict__ b2,
    float* __restrict__ out,
    int batch)
{
    // Masked-weight staging: only the live (block-diagonal) values.
    __shared__ float w0s[4096];   // effective W0: w0s[q] = W0[q, q/256]
    __shared__ float b0s[4096];
    __shared__ float b1s[2048];
    __shared__ float w2s[2048];   // w2s[i*128 + j*8 + h] = W2[i, j*128 + i*8 + h]
    __shared__ float b2s[16];

    const int t = threadIdx.x;

    // Contiguous biases via async DMA to LDS (ASYNCcnt path).
    async_stage_f32(b0, b0s, 4096 * 4, t);
    async_stage_f32(b1, b1s, 2048 * 4, t);

    // Gathered (masked) weights via regular loads.
    #pragma unroll
    for (int u = 0; u < 16; ++u) {
        int q = t * 16 + u;
        w0s[q] = W0[q * 16 + (q >> 8)];
    }
    #pragma unroll
    for (int u = 0; u < 8; ++u) {
        int r = t * 8 + u;
        int ii = r >> 7, jj = (r >> 3) & 15, hh = r & 7;
        w2s[r] = W2[ii * 2048 + jj * 128 + ii * 8 + hh];
    }
    if (t < 16) b2s[t] = b2[t];

    asm volatile("s_wait_asynccnt 0x0" ::: "memory");
    __syncthreads();

    const int wave = t >> 5;
    const int lid  = t & 31;
    const int m    = lid & 15;     // batch row within tile / D column index
    const int half = lid >> 4;     // A lane-half (K sub-range selector)
    const int tile = blockIdx.x * 8 + wave;
    const int row  = tile * 16 + m;
    if (row >= batch) return;

    // This lane's batch row (x is [B,16] row-major, 64B aligned rows).
    float xr[16];
    const float4* xv = reinterpret_cast<const float4*>(x + (size_t)row * 16);
    #pragma unroll
    for (int u = 0; u < 4; ++u) {
        float4 v = xv[u];
        xr[4*u+0] = v.x; xr[4*u+1] = v.y; xr[4*u+2] = v.z; xr[4*u+3] = v.w;
    }

    const int tileBase = tile * 16;

    // B-matrix lane roles for the 2-block packing:
    //  lanes 0..7   : K 0..15,  N=lid    -> W1 row of block jA
    //  lanes 24..31 : K 16..31, N=lid-16 -> W1 row of block jB
    //  lanes 8..23  : zero quadrants
    const bool loA = (lid < 8);
    const bool loB = (lid >= 24);
    const int  hB  = loA ? lid : (lid - 24);   // W1 row within its block

    for (int i = 0; i < OUT_DIM; ++i) {
        float acc[8];
        #pragma unroll
        for (int r = 0; r < 8; ++r) acc[r] = 0.f;

        for (int j = 0; j < IN_DIM; j += 2) {
            const int jA = j, jB = j + 1;
            const int kA = jA * 16 + i;
            const int kB = jB * 16 + i;

            // ---- A (16x32 bf16): half 0 slots 0..15 = block jA (g = s),
            //                      half 1 slots 0..15 = block jB (g = s).
            v16bf a;
            const int   jh = half ? jB : jA;
            const int   qA = jh * 256 + i * 16;
            const int   qB = qA + 8;
            const float xa = xr[jh], xb = xr[jh];
            #pragma unroll
            for (int s = 0; s < 8; ++s) {
                float va = fmaxf(fmaf(xa, w0s[qA + s], b0s[qA + s]), 0.f);
                float vb = fmaxf(fmaf(xb, w0s[qB + s], b0s[qB + s]), 0.f);
                a[s]     = (__bf16)va;
                a[8 + s] = (__bf16)vb;
            }

            // ---- B (32x16 bf16): block-diagonal 2-j packing.
            v16bf bm;
            #pragma unroll
            for (int s = 0; s < 16; ++s) bm[s] = (__bf16)0.f;
            if (loA || loB) {
                const int     kk   = loA ? kA : kB;
                const float*  wrow = W1 + (size_t)(8 * kk + hB) * 4096 + 16 * kk;
                const float4* w4   = reinterpret_cast<const float4*>(wrow);
                #pragma unroll
                for (int u = 0; u < 4; ++u) {
                    float4 v = w4[u];
                    bm[4*u+0] = (__bf16)v.x;
                    bm[4*u+1] = (__bf16)v.y;
                    bm[4*u+2] = (__bf16)v.z;
                    bm[4*u+3] = (__bf16)v.w;
                }
                if (j + 2 < IN_DIM) {  // gfx1250 global_prefetch_b8 for next pair
                    const int kn = kk + 32;   // same i, j += 2
                    __builtin_prefetch(W1 + (size_t)(8 * kn + hB) * 4096 + 16 * kn, 0, 1);
                }
            }

            // ---- C: bias b1 broadcast; every column live now.
            const float cv = (m < H1) ? b1s[kA * 8 + m] : b1s[kB * 8 + (m - 8)];
            v8f c;
            #pragma unroll
            for (int r = 0; r < 8; ++r) c[r] = cv;

            // ---- D = A x B + C  (v_wmma_f32_16x16x32_bf16, fully packed)
            c = __builtin_amdgcn_wmma_f32_16x16x32_bf16(
                    false, a, false, bm, (short)0, c, false, false);

            // ---- ReLU + fold output-layer weight; accumulate over j pairs.
            const float w2l = (m < H1) ? w2s[i * 128 + jA * 8 + m]
                                       : w2s[i * 128 + jB * 8 + (m - 8)];
            #pragma unroll
            for (int r = 0; r < 8; ++r)
                acc[r] += fmaxf(c[r], 0.f) * w2l;
        }

        // Reduce all 16 N-columns (= Σ over both packed j's and h) per half.
        #pragma unroll
        for (int r = 0; r < 8; ++r) {
            acc[r] += __shfl_xor(acc[r], 1, 32);
            acc[r] += __shfl_xor(acc[r], 2, 32);
            acc[r] += __shfl_xor(acc[r], 4, 32);
            acc[r] += __shfl_xor(acc[r], 8, 32);
        }

        if (lid == 0 || lid == 16) {
            const int m0 = half * 8;  // lane 0 -> rows 0..7, lane 16 -> rows 8..15
            #pragma unroll
            for (int r = 0; r < 8; ++r)
                out[(size_t)(tileBase + m0 + r) * OUT_DIM + i] = acc[r] + b2s[i];
        }
    }
}

extern "C" void kernel_launch(void* const* d_in, const int* in_sizes, int n_in,
                              void* d_out, int out_size, void* d_ws, size_t ws_size,
                              hipStream_t stream) {
    const float* x  = (const float*)d_in[0];
    const float* W0 = (const float*)d_in[1];
    const float* b0 = (const float*)d_in[2];
    const float* W1 = (const float*)d_in[3];
    const float* b1 = (const float*)d_in[4];
    const float* W2 = (const float*)d_in[5];
    const float* b2 = (const float*)d_in[6];
    float* out = (float*)d_out;

    const int batch  = in_sizes[0] / IN_DIM;   // 8192
    const int tiles  = batch / 16;             // 512 (one wave each)
    const int blocks = (tiles + 7) / 8;        // 64 blocks of 8 waves

    sparse_mlp_wmma<<<blocks, 256, 0, stream>>>(x, W0, b0, W1, b1, W2, b2, out, batch);
}
